// GINNet_9251359555641
// MI455X (gfx1250) — matrix operand order repaired and checked by reading, not compiled, not run on hardware
//
#include <hip/hip_runtime.h>

// ---------------------------------------------------------------------------
// GIN forward on MI455X (gfx1250, wave32).
// GEMMs use v_wmma_f32_16x16x32_bf16 (f32 accumulate) with 32x64 register
// blocking per wave (2 A-frags x 4 B-frags -> 8 WMMAs per 32-deep k-step,
// ~16 FLOP/byte). Weights pre-transposed to bf16 [Npad x Kpad32] so each
// lane's B fragment is one aligned 32-byte vector load. Activations f32 are
// converted to bf16 in-register in contiguous k order (k = 16*half + s), same as B.
// All activation buffers padded to M_P rows -> no row guards anywhere.
// ---------------------------------------------------------------------------

typedef __attribute__((ext_vector_type(16))) __bf16 v16bf;
typedef __attribute__((ext_vector_type(8)))  float  v8f;
typedef unsigned short ushort_t;

#define BN_EPS 1e-5f

#define N_NODES 50000
#define M_P     50176        // 196 * 256 (block covers 256 rows)
#define N_EDGES 200000
#define IN_DIM  64
#define EMB     300
#define EMB_P   320          // K padding (mult 32) and ld of h/z
#define EMB_PN  320          // N padding to mult of 64 for B tiles
#define HID     600
#define HID_P   608          // K padding (mult 32) and ld of mid
#define HID_PN  640          // N padding to mult of 64 for B tiles
#define N_GRAPHS 1024
#define N_CLASS 10

// -------------------------------- utility ----------------------------------

__global__ void fill_zero4(float4* p, size_t n4) {
  size_t i = (size_t)blockIdx.x * blockDim.x + threadIdx.x;
  size_t s = (size_t)gridDim.x * blockDim.x;
  float4 z = {0.f, 0.f, 0.f, 0.f};
  for (; i < n4; i += s) p[i] = z;
}

__global__ void copy4(float4* __restrict__ d, const float4* __restrict__ s, size_t n4) {
  size_t i = (size_t)blockIdx.x * blockDim.x + threadIdx.x;
  size_t st = (size_t)gridDim.x * blockDim.x;
  for (; i < n4; i += st) d[i] = s[i];
}

// W is row-major [K x N] f32; Bt is bf16 [Npad x Kpad], zero padded.
__global__ void convert_weight_t(const float* __restrict__ W, ushort_t* __restrict__ Bt,
                                 int K, int N, int Kpad, int Npad) {
  int idx = blockIdx.x * blockDim.x + threadIdx.x;
  int total = Kpad * Npad;
  if (idx >= total) return;
  int n = idx / Kpad, k = idx - n * Kpad;
  float v = (k < K && n < N) ? W[(size_t)k * N + n] : 0.f;
  __bf16 h = (__bf16)v;
  Bt[(size_t)n * Kpad + k] = __builtin_bit_cast(ushort_t, h);
}

// z[dst] += h[src] per edge (z must already contain h: GIN eps=0 sum).
__global__ void scatter_add_rows(const int* __restrict__ src, const int* __restrict__ dst,
                                 const float* __restrict__ h, float* __restrict__ z,
                                 int dim, int ld) {
  int e = blockIdx.x;
  const float* hr = h + (size_t)src[e] * ld;
  float*       zr = z + (size_t)dst[e] * ld;
  for (int f = threadIdx.x; f < dim; f += blockDim.x)
    atomicAdd(&zr[f], hr[f]);
}

// --------------------------- WMMA GEMM + BN --------------------------------
// A-fragment loader: lanes 0-15 hold K = {0..15}, lanes 16-31 hold
// K = {16..31} (contiguous, matching B); p pre-offset by lhalf*16.
__device__ inline v16bf load_a_bf16(const float* p) {
  float4 a0 = *(const float4*)(p);
  float4 a1 = *(const float4*)(p + 4);
  float4 a2 = *(const float4*)(p + 8);
  float4 a3 = *(const float4*)(p + 12);
  v16bf a;
  a[0]  = (__bf16)a0.x; a[1]  = (__bf16)a0.y; a[2]  = (__bf16)a0.z; a[3]  = (__bf16)a0.w;
  a[4]  = (__bf16)a1.x; a[5]  = (__bf16)a1.y; a[6]  = (__bf16)a1.z; a[7]  = (__bf16)a1.w;
  a[8]  = (__bf16)a2.x; a[9]  = (__bf16)a2.y; a[10] = (__bf16)a2.z; a[11] = (__bf16)a2.w;
  a[12] = (__bf16)a3.x; a[13] = (__bf16)a3.y; a[14] = (__bf16)a3.z; a[15] = (__bf16)a3.w;
  return a;
}

// C[Mp x N] = BN(A[Mp x K] * B[K x N] + bias) (+ReLU). A f32 (lda = K, mult
// of 32, rows padded), Bt bf16 transposed [n][k], ldbt = Kpad, Npad rows.
// One wave computes a 32x64 tile; 8 waves/block stacked in M (256 x 64).
__global__ __launch_bounds__(256) void wmma_gemm_bn(
    const float* __restrict__ A, int lda,
    const ushort_t* __restrict__ Bt, int ldbt,
    float* __restrict__ C, int ldc,
    int N, int K,
    const float* __restrict__ bias,
    const float* __restrict__ bng, const float* __restrict__ bnb,
    const float* __restrict__ bnm, const float* __restrict__ bnv,
    int relu)
{
  const int wave  = threadIdx.x >> 5;
  const int lane  = threadIdx.x & 31;
  const int m0    = blockIdx.x * 256 + wave * 32;   // rows m0..m0+31 (always in-range: padded)
  const int n0    = blockIdx.y * 64;                // cols n0..n0+63 (B rows padded to Npad)
  const int lhalf = lane >> 4;
  const int l15   = lane & 15;

  const float*    Arow0 = A + (size_t)(m0 + l15) * lda + lhalf * 16;
  const float*    Arow1 = Arow0 + (size_t)16 * lda;
  const ushort_t* Bc    = Bt + (size_t)(n0 + l15) * ldbt + lhalf * 16;
  const size_t    bstep = (size_t)16 * ldbt;        // next 16-column group

  v8f acc[2][4];
#pragma unroll
  for (int i = 0; i < 2; ++i)
#pragma unroll
    for (int j = 0; j < 4; ++j) acc[i][j] = (v8f){};

  for (int k0 = 0; k0 < K; k0 += 32) {
    v16bf a0 = load_a_bf16(Arow0 + k0);
    v16bf a1 = load_a_bf16(Arow1 + k0);
    v16bf b0 = *(const v16bf*)(const void*)(Bc + k0);
    v16bf b1 = *(const v16bf*)(const void*)(Bc + bstep + k0);
    v16bf b2 = *(const v16bf*)(const void*)(Bc + 2 * bstep + k0);
    v16bf b3 = *(const v16bf*)(const void*)(Bc + 3 * bstep + k0);
    acc[0][0] = __builtin_amdgcn_wmma_f32_16x16x32_bf16(false, a0, false, b0, (short)0, acc[0][0], false, false);
    acc[0][1] = __builtin_amdgcn_wmma_f32_16x16x32_bf16(false, a0, false, b1, (short)0, acc[0][1], false, false);
    acc[0][2] = __builtin_amdgcn_wmma_f32_16x16x32_bf16(false, a0, false, b2, (short)0, acc[0][2], false, false);
    acc[0][3] = __builtin_amdgcn_wmma_f32_16x16x32_bf16(false, a0, false, b3, (short)0, acc[0][3], false, false);
    acc[1][0] = __builtin_amdgcn_wmma_f32_16x16x32_bf16(false, a1, false, b0, (short)0, acc[1][0], false, false);
    acc[1][1] = __builtin_amdgcn_wmma_f32_16x16x32_bf16(false, a1, false, b1, (short)0, acc[1][1], false, false);
    acc[1][2] = __builtin_amdgcn_wmma_f32_16x16x32_bf16(false, a1, false, b2, (short)0, acc[1][2], false, false);
    acc[1][3] = __builtin_amdgcn_wmma_f32_16x16x32_bf16(false, a1, false, b3, (short)0, acc[1][3], false, false);
  }

  // Fused epilogue: y = (dot + bias) * S + (b - m*S), S = g * rsqrt(v + eps)
#pragma unroll
  for (int j = 0; j < 4; ++j) {
    const int ncol = n0 + j * 16 + l15;
    if (ncol < N) {
      float S  = bng[ncol] * rsqrtf(bnv[ncol] + BN_EPS);
      float sh = bnb[ncol] - bnm[ncol] * S;
      float bi = bias[ncol];
#pragma unroll
      for (int i = 0; i < 2; ++i) {
        // D layout: lane covers n = l15, rows m = i*16 + lhalf*8 + v
        float* Cc = C + (size_t)(m0 + i * 16 + lhalf * 8) * ldc + ncol;
#pragma unroll
        for (int v = 0; v < 8; ++v) {
          float y = (acc[i][j][v] + bi) * S + sh;
          if (relu) y = fmaxf(y, 0.f);
          Cc[(size_t)v * ldc] = y;
        }
      }
    }
  }
}

// ------------------------------ pooling ------------------------------------

__global__ void pool_kernel(const float* __restrict__ h, const int* __restrict__ batch,
                            float* __restrict__ pooled, float* __restrict__ counts,
                            int dim, int ld) {
  int i = blockIdx.x;
  int g = batch[i];
  const float* hr = h + (size_t)i * ld;
  float* pr = pooled + (size_t)g * dim;
  for (int f = threadIdx.x; f < dim; f += blockDim.x)
    atomicAdd(&pr[f], hr[f]);
  if (threadIdx.x == 0) atomicAdd(&counts[g], 1.0f);
}

__global__ void classifier_kernel(const float* __restrict__ pooled, const float* __restrict__ counts,
                                  const float* __restrict__ W, const float* __restrict__ b,
                                  float* __restrict__ out, int dim) {
  int idx = blockIdx.x * blockDim.x + threadIdx.x;
  if (idx >= N_GRAPHS * N_CLASS) return;
  int g = idx / N_CLASS, c = idx - g * N_CLASS;
  float inv = 1.0f / fmaxf(counts[g], 1.0f);
  const float* pr = pooled + (size_t)g * dim;
  float s = 0.f;
  for (int f = 0; f < dim; ++f) s += pr[f] * W[(size_t)f * N_CLASS + c];
  out[idx] = s * inv + b[c];
}

// ------------------------------- launcher ----------------------------------

extern "C" void kernel_launch(void* const* d_in, const int* in_sizes, int n_in,
                              void* d_out, int out_size, void* d_ws, size_t ws_size,
                              hipStream_t stream) {
  const float* x      = (const float*)d_in[0];
  const int*   eidx   = (const int*)d_in[1];
  const int*   batch  = (const int*)d_in[2];
  const float* l1_W1  = (const float*)d_in[3];
  const float* l1_b1  = (const float*)d_in[4];
  const float* l1_bng = (const float*)d_in[5];
  const float* l1_bnb = (const float*)d_in[6];
  const float* l1_bnm = (const float*)d_in[7];
  const float* l1_bnv = (const float*)d_in[8];
  const float* l1_W2  = (const float*)d_in[9];
  const float* l1_b2  = (const float*)d_in[10];
  const float* l1_bog = (const float*)d_in[11];
  const float* l1_bob = (const float*)d_in[12];
  const float* l1_bom = (const float*)d_in[13];
  const float* l1_bov = (const float*)d_in[14];
  const float* Ws1    = (const float*)d_in[15];
  const float* bs1    = (const float*)d_in[16];
  const float* bn_g   = (const float*)d_in[17];
  const float* bn_b   = (const float*)d_in[18];
  const float* bn_m   = (const float*)d_in[19];
  const float* bn_v   = (const float*)d_in[20];
  const float* Ws2    = (const float*)d_in[21];
  const float* bs2    = (const float*)d_in[22];
  const float* bno_g  = (const float*)d_in[23];
  const float* bno_b  = (const float*)d_in[24];
  const float* bno_m  = (const float*)d_in[25];
  const float* bno_v  = (const float*)d_in[26];
  const float* clf_W  = (const float*)d_in[27];
  const float* clf_b  = (const float*)d_in[28];
  float* out = (float*)d_out;

  const int* src = eidx;              // edge_index[0]
  const int* dst = eidx + N_EDGES;    // edge_index[1]

  // ---- workspace carve (256B aligned) ----
  char* ws = (char*)d_ws;
  auto alloc = [&](size_t bytes) -> char* {
    char* p = ws; ws += (bytes + 255) & ~(size_t)255; return p;
  };
  float*    hbuf  = (float*)alloc((size_t)M_P * EMB_P * 4);
  float*    zbuf  = (float*)alloc((size_t)M_P * EMB_P * 4);
  float*    mid   = (float*)alloc((size_t)M_P * HID_P * 4);
  ushort_t* W1tL1 = (ushort_t*)alloc((size_t)HID_PN * IN_DIM * 2);
  ushort_t* W2tL1 = (ushort_t*)alloc((size_t)EMB_PN * HID_P * 2);
  ushort_t* Ws1t  = (ushort_t*)alloc((size_t)4 * HID_PN * EMB_P * 2);
  ushort_t* Ws2t  = (ushort_t*)alloc((size_t)4 * EMB_PN * HID_P * 2);
  float*    pooled= (float*)alloc((size_t)N_GRAPHS * EMB * 4);
  float*    counts= (float*)alloc((size_t)N_GRAPHS * 4);

  // ---- zero activations (hbuf/zbuf/mid contiguous) and pooled/counts ----
  {
    size_t n4 = ((size_t)M_P * EMB_P * 2 + (size_t)M_P * HID_P) / 4;
    fill_zero4<<<2048, 256, 0, stream>>>((float4*)hbuf, n4);
    size_t p4 = ((size_t)N_GRAPHS * EMB + N_GRAPHS) / 4;
    fill_zero4<<<512, 256, 0, stream>>>((float4*)pooled, p4);
  }

  // ---- weight transpose + f32->bf16 (cheap, every call: deterministic) ----
  {
    int t = HID_PN * IN_DIM;
    convert_weight_t<<<(t + 255) / 256, 256, 0, stream>>>(l1_W1, W1tL1, IN_DIM, HID, IN_DIM, HID_PN);
    t = EMB_PN * HID_P;
    convert_weight_t<<<(t + 255) / 256, 256, 0, stream>>>(l1_W2, W2tL1, HID, EMB, HID_P, EMB_PN);
    for (int i = 0; i < 4; ++i) {
      t = HID_PN * EMB_P;
      convert_weight_t<<<(t + 255) / 256, 256, 0, stream>>>(
          Ws1 + (size_t)i * EMB * HID, Ws1t + (size_t)i * HID_PN * EMB_P, EMB, HID, EMB_P, HID_PN);
      t = EMB_PN * HID_P;
      convert_weight_t<<<(t + 255) / 256, 256, 0, stream>>>(
          Ws2 + (size_t)i * HID * EMB, Ws2t + (size_t)i * EMB_PN * HID_P, HID, EMB, HID_P, EMB_PN);
    }
  }

  const dim3 gemmBlk(256);
  const dim3 grid1(M_P / 256, HID_PN / 64);   // (196, 10) for N=600
  const dim3 grid2(M_P / 256, EMB_PN / 64);   // (196, 5)  for N=300

  // ================= layer 1 (IN_DIM -> EMB) =================
  copy4<<<2048, 256, 0, stream>>>((float4*)zbuf, (const float4*)x, (size_t)N_NODES * IN_DIM / 4);
  scatter_add_rows<<<N_EDGES, 64, 0, stream>>>(src, dst, x, zbuf, IN_DIM, IN_DIM);
  wmma_gemm_bn<<<grid1, gemmBlk, 0, stream>>>(zbuf, IN_DIM, W1tL1, IN_DIM, mid, HID_P,
                                              HID, IN_DIM,
                                              l1_b1, l1_bng, l1_bnb, l1_bnm, l1_bnv, 1);
  wmma_gemm_bn<<<grid2, gemmBlk, 0, stream>>>(mid, HID_P, W2tL1, HID_P, hbuf, EMB_P,
                                              EMB, HID_P,
                                              l1_b2, l1_bog, l1_bob, l1_bom, l1_bov, 1);

  // ================= layers 2..5 (EMB -> EMB) =================
  for (int i = 0; i < 4; ++i) {
    copy4<<<2048, 256, 0, stream>>>((float4*)zbuf, (const float4*)hbuf, (size_t)M_P * EMB_P / 4);
    scatter_add_rows<<<N_EDGES, 64, 0, stream>>>(src, dst, hbuf, zbuf, EMB, EMB_P);
    wmma_gemm_bn<<<grid1, gemmBlk, 0, stream>>>(zbuf, EMB_P, Ws1t + (size_t)i * HID_PN * EMB_P, EMB_P,
                                                mid, HID_P, HID, EMB_P,
                                                bs1 + (size_t)i * HID,
                                                bn_g + (size_t)i * HID, bn_b + (size_t)i * HID,
                                                bn_m + (size_t)i * HID, bn_v + (size_t)i * HID, 1);
    wmma_gemm_bn<<<grid2, gemmBlk, 0, stream>>>(mid, HID_P, Ws2t + (size_t)i * EMB_PN * HID_P, HID_P,
                                                hbuf, EMB_P, EMB, HID_P,
                                                bs2 + (size_t)i * EMB,
                                                bno_g + (size_t)i * EMB, bno_b + (size_t)i * EMB,
                                                bno_m + (size_t)i * EMB, bno_v + (size_t)i * EMB,
                                                (i != 3) ? 1 : 0);
  }

  // ================= mean pool + classifier =================
  pool_kernel<<<N_NODES, 64, 0, stream>>>(hbuf, batch, pooled, counts, EMB, EMB_P);
  classifier_kernel<<<(N_GRAPHS * N_CLASS + 255) / 256, 256, 0, stream>>>(
      pooled, counts, clf_W, clf_b, out, EMB);

  (void)in_sizes; (void)n_in; (void)out_size; (void)ws_size;
}
